// CTM_34574486733138
// MI455X (gfx1250) — matrix operand order repaired and checked by reading, not compiled, not run on hardware
//
#include <hip/hip_runtime.h>
#include <hip/hip_bf16.h>
#include <math.h>

// ---------------- problem constants ----------------
#define BB 4
#define NN 3072
#define CC 512
#define CL 768          // CLUSTER_NUM
#define KNN 5

typedef __attribute__((ext_vector_type(16))) __bf16 v16bf;
typedef __attribute__((ext_vector_type(8)))  float  v8f;

// ---------------- workspace layout (bytes) ----------------
static constexpr size_t OFF_XB   = 0;                                    // ushort  B*N*C  (bf16 copy of x)
static constexpr size_t SZ_XB    = (size_t)BB * NN * CC * 2;             // 12,582,912
static constexpr size_t OFF_SQ   = OFF_XB + SZ_XB;                       // float   B*N
static constexpr size_t OFF_DEN  = OFF_SQ  + (size_t)BB * NN * 4;        // float   B*N
static constexpr size_t OFF_SCR  = OFF_DEN + (size_t)BB * NN * 4;        // float   B*N (score)
static constexpr size_t OFF_DMAX = OFF_SCR + (size_t)BB * NN * 4;        // uint    B (float bits)
static constexpr size_t OFF_IDN  = OFF_DMAX + 256;                       // int     B*CL (index_down)
static constexpr size_t OFF_IDC  = OFF_IDN + (size_t)BB * CL * 4;        // int     B*N (idx_cluster)
static constexpr size_t OFF_CNT  = OFF_IDC + (size_t)BB * NN * 4;        // float   B*CL (all_weight counts)
static constexpr size_t OFF_DIST = ((OFF_CNT + (size_t)BB * CL * 4 + 255) / 256) * 256; // float B*N*N

// ---------------- output layout (float elems) ----------------
static constexpr size_t OUT_XM  = 0;                                     // B*CL*C
static constexpr size_t OUT_AGG = (size_t)BB * CL * CC;                  // B*N
static constexpr size_t OUT_IDT = OUT_AGG + (size_t)BB * NN;             // B*N (int as float)
static constexpr size_t OUT_IDC = OUT_IDT + (size_t)BB * NN;             // B*N (int as float)

__device__ __forceinline__ float hash_uniform(unsigned s) {
    s ^= s >> 16; s *= 0x7feb352dU; s ^= s >> 15; s *= 0x846ca68bU; s ^= s >> 16;
    return (float)(s & 0xFFFFFFu) * (1.0f / 16777216.0f);
}

// low 32 bits of a flat LDS pointer = LDS byte address for async-to-LDS ops
__device__ __forceinline__ unsigned lds_off32(const void* p) {
    return (unsigned)(unsigned long long)p;
}

// CDNA5 async DMA: 16 bytes global -> LDS, tracked by ASYNCcnt (no VGPR round-trip)
__device__ __forceinline__ void async_cp16(unsigned lds_addr, const void* gp) {
    asm volatile("global_load_async_to_lds_b128 %0, %1, off"
                 :: "v"(lds_addr), "v"((unsigned long long)gp)
                 : "memory");
}
__device__ __forceinline__ void wait_async0() {
    asm volatile("s_wait_asynccnt 0" ::: "memory");
}

// ---- 0: zero x_merged region of out, counts, distmax -----------------------
__global__ void tcf_init(float* out, float* counts, unsigned* dmax) {
    size_t g = (size_t)blockIdx.x * blockDim.x + threadIdx.x;
    size_t tot = (size_t)BB * CL * CC;
    for (size_t i = g; i < tot; i += (size_t)gridDim.x * blockDim.x) out[OUT_XM + i] = 0.0f;
    if (g < BB * CL) counts[g] = 0.0f;
    if (g < BB) dmax[g] = 0u;
}

// ---- 1: convert x -> bf16, compute row squared norms (1 wave / row) --------
__global__ __launch_bounds__(256) void tcf_prep(const float* __restrict__ x,
                                                unsigned short* __restrict__ xb,
                                                float* __restrict__ sq) {
    int row  = blockIdx.x * 8 + (threadIdx.x >> 5);   // B*N rows
    int lane = threadIdx.x & 31;
    const float* xr = x + (size_t)row * CC;
    unsigned short* br = xb + (size_t)row * CC;
    float s = 0.0f;
    for (int c = lane; c < CC; c += 32) {
        float v = xr[c];
        s += v * v;
        unsigned u = __float_as_uint(v);
        u += 0x7FFFu + ((u >> 16) & 1u);              // RNE to bf16
        br[c] = (unsigned short)(u >> 16);
    }
    for (int off = 16; off > 0; off >>= 1) s += __shfl_xor(s, off, 32);
    if (lane == 0) sq[row] = s;
}

// ---- 2: dist GEMM via v_wmma_f32_16x16x32_bf16, async double-buffered ------
// dist[b,i,j] = sqrt(max(sq_i + sq_j - 2*dot, 0)) / sqrt(C)
#define MT 128
#define NT 128
#define KC 32
__global__ __launch_bounds__(256) void tcf_dist(const unsigned short* __restrict__ xb,
                                                const float* __restrict__ sq,
                                                float* __restrict__ dist) {
    const int b  = blockIdx.z;
    const int ti = blockIdx.y;
    const int tj = blockIdx.x;
    const unsigned short* Xb = xb + (size_t)b * NN * CC;

    __shared__ unsigned short As[2][MT][KC];   // 16 KB (double-buffered)
    __shared__ unsigned short Bs[2][NT][KC];   // 16 KB

    const int tid  = threadIdx.x;
    const int lane = tid & 31;
    const int wid  = tid >> 5;              // 8 waves
    const int wm = (wid >> 2) * 64;         // 0 / 64
    const int wn = (wid & 3) * 32;          // 0..96

    // this thread's two 16B chunks: chunk c -> row = c>>2, col = (c&3)*8
    const int r0 = tid >> 2,           c0 = (tid & 3) * 8;
    const int r1 = (tid + 256) >> 2,   c1 = ((tid + 256) & 3) * 8;

    const unsigned aL = lds_off32(&As[0][0][0]);
    const unsigned bL = lds_off32(&Bs[0][0][0]);
    const unsigned bufStride = MT * KC * 2;           // bytes per buffer

    v8f acc[4][2];
    for (int a = 0; a < 4; ++a)
        for (int bnn = 0; bnn < 2; ++bnn) acc[a][bnn] = (v8f){};

    union Frag { uint4 q[2]; v16bf v; };

    // stage K-tile k0 into buffer `buf` (4 async 16B DMAs per thread)
    auto stage = [&](int buf, int k0) {
        unsigned ab = aL + buf * bufStride;
        unsigned bb = bL + buf * bufStride;
        async_cp16(ab + (unsigned)(r0 * KC + c0) * 2,
                   &Xb[(size_t)(ti * MT + r0) * CC + k0 + c0]);
        async_cp16(ab + (unsigned)(r1 * KC + c1) * 2,
                   &Xb[(size_t)(ti * MT + r1) * CC + k0 + c1]);
        async_cp16(bb + (unsigned)(r0 * KC + c0) * 2,
                   &Xb[(size_t)(tj * NT + r0) * CC + k0 + c0]);
        async_cp16(bb + (unsigned)(r1 * KC + c1) * 2,
                   &Xb[(size_t)(tj * NT + r1) * CC + k0 + c1]);
    };

    stage(0, 0);
    wait_async0();
    __syncthreads();

    for (int k0 = 0; k0 < CC; k0 += KC) {
        const int buf = (k0 / KC) & 1;
        if (k0 + KC < CC) {
            stage(buf ^ 1, k0 + KC);                    // overlap DMA with WMMA
            if (k0 + 2 * KC < CC) {                     // prefetch tile after next
                __builtin_prefetch(&Xb[(size_t)(ti * MT + r0) * CC + k0 + 2 * KC + c0], 0, 0);
                __builtin_prefetch(&Xb[(size_t)(tj * NT + r0) * CC + k0 + 2 * KC + c0], 0, 0);
            }
        }

        // B fragments (32x16 bf16): lane<16 -> col=lane,K 0..7 & 16..23 ; lane>=16 -> K 8..15 & 24..31
        Frag bf[2];
        {
            const int kb = (lane < 16) ? 0 : 8;
            for (int sn = 0; sn < 2; ++sn) {
                int col = wn + sn * 16 + (lane & 15);
                bf[sn].q[0] = *(const uint4*)&Bs[buf][col][kb];
                bf[sn].q[1] = *(const uint4*)&Bs[buf][col][kb + 16];
            }
        }
        // A fragments (16x32 bf16): lane<16 -> K 0..7 & 16..23 ; lane>=16 -> K 8..15 & 24..31
        {
            const int kb = (lane < 16) ? 0 : 8;
            for (int sm = 0; sm < 4; ++sm) {
                Frag af;
                int row = wm + sm * 16 + (lane & 15);
                af.q[0] = *(const uint4*)&As[buf][row][kb];
                af.q[1] = *(const uint4*)&As[buf][row][kb + 16];
                for (int sn = 0; sn < 2; ++sn)
                    acc[sm][sn] = __builtin_amdgcn_wmma_f32_16x16x32_bf16(
                        false, af.v, false, bf[sn].v, (short)0, acc[sm][sn],
                        false, false);
            }
        }

        wait_async0();        // own wave's DMAs into buf^1 done
        __syncthreads();      // all waves' DMAs visible; buf free for reuse
    }

    // epilogue: D layout -> lanes 0-15: n=lane, m=r ; lanes 16-31: n=lane-16, m=r+8
    const float rs = 0.044194173824159216f;   // 1/sqrt(512)
    const float* sqb = sq + (size_t)b * NN;
    float* Db = dist + (size_t)b * NN * NN;
    const int nloc = lane & 15;
    const int mofs = (lane < 16) ? 0 : 8;
    for (int sm = 0; sm < 4; ++sm)
        for (int sn = 0; sn < 2; ++sn) {
            int gj = tj * NT + wn + sn * 16 + nloc;
            float sj = sqb[gj];
            for (int r = 0; r < 8; ++r) {
                int gi = ti * MT + wm + sm * 16 + r + mofs;
                float v = sqb[gi] + sj - 2.0f * acc[sm][sn][r];
                Db[(size_t)gi * NN + gj] = sqrtf(fmaxf(v, 0.0f)) * rs;
            }
        }
}

// ---- 3: per-row 5-NN -> density (1 wave / row) -----------------------------
__global__ __launch_bounds__(256) void tcf_density(const float* __restrict__ dist,
                                                   float* __restrict__ density) {
    int row  = blockIdx.x * 8 + (threadIdx.x >> 5);
    int lane = threadIdx.x & 31;
    int wid  = threadIdx.x >> 5;
    const float* dr = dist + (size_t)row * NN;
    float best[KNN];
    for (int t = 0; t < KNN; ++t) best[t] = 3.4e38f;
    for (int j = lane; j < NN; j += 32) {
        float d = dr[j];
        if (d < best[KNN - 1]) {
            best[KNN - 1] = d;
            for (int t = KNN - 1; t > 0; --t)
                if (best[t] < best[t - 1]) { float tmp = best[t]; best[t] = best[t - 1]; best[t - 1] = tmp; }
                else break;
        }
    }
    __shared__ float buf[8][32 * KNN];
    for (int t = 0; t < KNN; ++t) buf[wid][lane * KNN + t] = best[t];
    __syncthreads();
    if (lane == 0) {
        float sel[KNN];
        for (int t = 0; t < KNN; ++t) sel[t] = 3.4e38f;
        for (int i = 0; i < 32 * KNN; ++i) {
            float d = buf[wid][i];
            if (d < sel[KNN - 1]) {
                sel[KNN - 1] = d;
                for (int t = KNN - 1; t > 0; --t)
                    if (sel[t] < sel[t - 1]) { float tmp = sel[t]; sel[t] = sel[t - 1]; sel[t - 1] = tmp; }
                    else break;
            }
        }
        float m = 0.0f;
        for (int t = 0; t < KNN; ++t) m += sel[t] * sel[t];
        m *= (1.0f / KNN);
        density[row] = expf(-m) + hash_uniform(42u * 2654435761u + (unsigned)row) * 1e-6f;
    }
}

// ---- 4: per-batch max(dist) ------------------------------------------------
__global__ __launch_bounds__(256) void tcf_dmax(const float* __restrict__ dist,
                                                unsigned* __restrict__ dmax) {
    const int chunks_per_b = (NN * NN) / 8192;
    int chunk = blockIdx.x;
    int b = chunk / chunks_per_b;
    size_t base = (size_t)b * NN * NN + (size_t)(chunk % chunks_per_b) * 8192;
    float m = 0.0f;
    for (int i = threadIdx.x; i < 8192; i += 256) m = fmaxf(m, dist[base + i]);
    for (int off = 16; off > 0; off >>= 1) m = fmaxf(m, __shfl_xor(m, off, 32));
    __shared__ float s[8];
    if ((threadIdx.x & 31) == 0) s[threadIdx.x >> 5] = m;
    __syncthreads();
    if (threadIdx.x == 0) {
        for (int w = 1; w < 8; ++w) m = fmaxf(m, s[w]);
        atomicMax(&dmax[b], __float_as_uint(m));   // dist >= 0: bit order == value order
    }
}

// ---- 5: masked min over higher-density tokens -> score (1 wave / row) ------
__global__ __launch_bounds__(256) void tcf_score(const float* __restrict__ dist,
                                                 const float* __restrict__ density,
                                                 const unsigned* __restrict__ dmax,
                                                 float* __restrict__ score) {
    int row  = blockIdx.x * 8 + (threadIdx.x >> 5);
    int lane = threadIdx.x & 31;
    int b = row / NN;
    const float* dr = dist + (size_t)row * NN;
    const float* db = density + (size_t)b * NN;
    float di = density[row];
    float mx = __uint_as_float(dmax[b]);
    float m = mx;
    for (int j = lane; j < NN; j += 32) {
        float v = (db[j] > di) ? dr[j] : mx;
        m = fminf(m, v);
    }
    for (int off = 16; off > 0; off >>= 1) m = fminf(m, __shfl_xor(m, off, 32));
    if (lane == 0) score[row] = m * di;
}

// ---- 6: per-batch top-768 (descending bitonic sort of 4096 in LDS) ---------
__global__ __launch_bounds__(1024) void tcf_topk(const float* __restrict__ score,
                                                 int* __restrict__ index_down) {
    int b = blockIdx.x;
    __shared__ float key[4096];
    __shared__ int   idx[4096];
    for (int i = threadIdx.x; i < 4096; i += 1024) {
        key[i] = (i < NN) ? score[(size_t)b * NN + i] : -3.4e38f;
        idx[i] = i;
    }
    __syncthreads();
    for (int k = 2; k <= 4096; k <<= 1)
        for (int j = k >> 1; j > 0; j >>= 1) {
            for (int i = threadIdx.x; i < 4096; i += 1024) {
                int p = i ^ j;
                if (p > i) {
                    bool up = ((i & k) == 0);          // descending sort
                    bool sw = up ? (key[i] < key[p]) : (key[i] > key[p]);
                    if (sw) {
                        float tk = key[i]; key[i] = key[p]; key[p] = tk;
                        int   ti2 = idx[i]; idx[i] = idx[p]; idx[p] = ti2;
                    }
                }
            }
            __syncthreads();
        }
    for (int c = threadIdx.x; c < CL; c += 1024) index_down[b * CL + c] = idx[c];
}

// ---- 7: assign each token to nearest center --------------------------------
__global__ __launch_bounds__(256) void tcf_assign(const float* __restrict__ dist,
                                                  const int* __restrict__ index_down,
                                                  int* __restrict__ idx_cluster) {
    int g = blockIdx.x * 256 + threadIdx.x;   // B*N tokens, 256 | N -> uniform b per block
    int b = g / NN, j = g % NN;
    __shared__ int cidx[CL];
    for (int c = threadIdx.x; c < CL; c += 256) cidx[c] = index_down[b * CL + c];
    __syncthreads();
    const float* Db = dist + (size_t)b * NN * NN;
    float best = 3.4e38f; int bi = 0;
    for (int c = 0; c < CL; ++c) {
        float d = Db[(size_t)cidx[c] * NN + j];
        if (d < best) { best = d; bi = c; }
    }
    idx_cluster[g] = bi;
}

// ---- 8: centers map to their own cluster id --------------------------------
__global__ void tcf_fixcenters(const int* __restrict__ index_down,
                               int* __restrict__ idx_cluster) {
    int g = blockIdx.x * blockDim.x + threadIdx.x;   // B*CL
    if (g >= BB * CL) return;
    int b = g / CL, c = g % CL;
    idx_cluster[b * NN + index_down[g]] = c;
}

// ---- 9: cluster token counts (all_weight) ----------------------------------
__global__ void tcf_counts(const int* __restrict__ idx_cluster,
                           float* __restrict__ counts) {
    int g = blockIdx.x * blockDim.x + threadIdx.x;   // B*N
    if (g >= BB * NN) return;
    int b = g / NN;
    atomicAdd(&counts[b * CL + idx_cluster[g]], 1.0f);
}

// ---- 10: merge tokens + aux outputs (1 block / token) ----------------------
__global__ __launch_bounds__(256) void tcf_merge(const float* __restrict__ x,
                                                 const int* __restrict__ idx_token,
                                                 const float* __restrict__ agg,
                                                 const int* __restrict__ idx_cluster,
                                                 const float* __restrict__ counts,
                                                 float* __restrict__ out) {
    int g = blockIdx.x;                    // B*N
    int b = g / NN;
    int cl = idx_cluster[g];
    float w = 1.0f / (counts[b * CL + cl] + 1e-6f);
    const float* xr = x + (size_t)g * CC;
    float* xm = out + OUT_XM + ((size_t)(b * CL + cl)) * CC;
    for (int c = threadIdx.x; c < CC; c += 256) atomicAdd(&xm[c], xr[c] * w);
    if (threadIdx.x == 0) {
        int it  = idx_token[g];
        int itn = idx_cluster[b * NN + it];
        float wt = 1.0f / (counts[b * CL + itn] + 1e-6f);
        out[OUT_AGG + g] = agg[g] * wt;
        out[OUT_IDT + g] = (float)itn;
        out[OUT_IDC + g] = (float)cl;
    }
}

extern "C" void kernel_launch(void* const* d_in, const int* in_sizes, int n_in,
                              void* d_out, int out_size, void* d_ws, size_t ws_size,
                              hipStream_t stream) {
    const float* x         = (const float*)d_in[0];
    const int*   idx_token = (const int*)d_in[1];
    const float* agg       = (const float*)d_in[2];
    float* out = (float*)d_out;

    char* ws = (char*)d_ws;
    unsigned short* xb   = (unsigned short*)(ws + OFF_XB);
    float*    sq         = (float*)(ws + OFF_SQ);
    float*    density    = (float*)(ws + OFF_DEN);
    float*    score      = (float*)(ws + OFF_SCR);
    unsigned* dmax       = (unsigned*)(ws + OFF_DMAX);
    int*      index_down = (int*)(ws + OFF_IDN);
    int*      idx_clu    = (int*)(ws + OFF_IDC);
    float*    counts     = (float*)(ws + OFF_CNT);
    float*    dist       = (float*)(ws + OFF_DIST);

    tcf_init<<<2048, 256, 0, stream>>>(out, counts, dmax);
    tcf_prep<<<(BB * NN) / 8, 256, 0, stream>>>(x, xb, sq);
    {
        dim3 grid(NN / NT, NN / MT, BB);     // 24 x 24 x 4
        tcf_dist<<<grid, 256, 0, stream>>>(xb, sq, dist);
    }
    tcf_density<<<(BB * NN) / 8, 256, 0, stream>>>(dist, density);
    tcf_dmax<<<BB * ((NN * NN) / 8192), 256, 0, stream>>>(dist, dmax);
    tcf_score<<<(BB * NN) / 8, 256, 0, stream>>>(dist, density, dmax, score);
    tcf_topk<<<BB, 1024, 0, stream>>>(score, index_down);
    tcf_assign<<<(BB * NN) / 256, 256, 0, stream>>>(dist, index_down, idx_clu);
    tcf_fixcenters<<<(BB * CL + 255) / 256, 256, 0, stream>>>(index_down, idx_clu);
    tcf_counts<<<(BB * NN + 255) / 256, 256, 0, stream>>>(idx_clu, counts);
    tcf_merge<<<BB * NN, 256, 0, stream>>>(x, idx_token, agg, idx_clu, counts, out);
}
